// GTransBlock_38929583571410
// MI455X (gfx1250) — matrix operand order repaired and checked by reading, not compiled, not run on hardware
//
#include <hip/hip_runtime.h>

#define IN_C   128
#define OUT_C  64
#define HC     256      // HEADS*OUT_C
#define FFN_H  256
#define SH     0.70710678f
#define LN_EPS 1e-7f

typedef __attribute__((ext_vector_type(16))) __bf16 v16bf;
typedef __attribute__((ext_vector_type(8)))  float  v8f;

// ---------- helpers ----------
__device__ __forceinline__ unsigned short f2bf(float f) {
    unsigned u = __float_as_uint(f);
    u += 0x7FFFu + ((u >> 16) & 1u);          // round-to-nearest-even
    return (unsigned short)(u >> 16);
}

// 16-bit A-matrix 16x32 lane layout (ISA 7.12.2): given row m (0..15) and
// local k (0..31), return (lane, elem) inside the 512-element fragment.
__device__ __forceinline__ void a_pos(int mrow, int kl, int& lane, int& e) {
    int hi   = (kl >= 16) ? 4 : 0;
    int r    = kl & 15;
    int half = r >> 3;
    int w    = r & 7;
    lane = half * 16 + mrow;
    e    = 2 * (hi + (w >> 1)) + (w & 1);
}

// one 16x16x32 bf16 WMMA step; fragments are 512 bf16 = [lane][16] contiguous
__device__ __forceinline__ v8f wmma_step(const unsigned short* aFrag,
                                         const unsigned short* bFrag, v8f c) {
    int lane = threadIdx.x & 31;
    v16bf a = *reinterpret_cast<const v16bf*>(aFrag + lane * 16);
    v16bf b = *reinterpret_cast<const v16bf*>(bFrag + lane * 16);
    return __builtin_amdgcn_wmma_f32_16x16x32_bf16(false, a, false, b,
                                                   (short)0, c, false, false);
}

// async-stage `chunks16` 16-byte chunks from global `g` into LDS at byte
// offset lds_base (per-thread strided by blockDim). ASYNCcnt-tracked.
__device__ __forceinline__ void async_stage(const unsigned short* g,
                                            unsigned lds_base, int chunks16,
                                            int tid, int nthreads) {
    for (int c = tid; c < chunks16; c += nthreads) {
        const unsigned short* ga = g + (size_t)c * 8;     // 16 bytes
        unsigned lofs = lds_base + c * 16;
        asm volatile("global_load_async_to_lds_b128 %0, %1, off"
                     :: "v"(lofs), "v"(ga) : "memory");
    }
}

// ---------- 1) weight packing: W (Nout x K row-major) -> B fragments ----------
// B-matrix 32x16 layout: lane = ((k/8)%2)*16 + n, elem = (k/16)*8 + k%8
__global__ void k_pack_w(const float* __restrict__ W, unsigned short* __restrict__ pack,
                         int NT, int KT, int K) {
    int t = blockIdx.x * blockDim.x + threadIdx.x;
    int total = NT * KT * 512;
    if (t >= total) return;
    int frag = t >> 9, within = t & 511;
    int lane = within >> 4, e = within & 15;
    int nt = frag / KT, kt = frag - nt * KT;
    int n  = nt * 16 + (lane & 15);
    int kl = (lane >> 4) * 8 + (e & 7) + ((e >> 3) << 4);
    pack[t] = f2bf(W[(size_t)n * K + kt * 32 + kl]);
}

// ---------- 2) LN(x) and pack x,h as bf16 A fragments (K=128 -> KT=4) ----------
__global__ void k_ln1_pack(const float* __restrict__ x, const float* __restrict__ g,
                           const float* __restrict__ beta,
                           unsigned short* __restrict__ xpack,
                           unsigned short* __restrict__ hpack, int N) {
    int wave = (blockIdx.x * blockDim.x + threadIdx.x) >> 5;
    int lane = threadIdx.x & 31;
    if (wave >= N) return;
    const float4 v = *reinterpret_cast<const float4*>(x + (size_t)wave * IN_C + lane * 4);
    float s  = v.x + v.y + v.z + v.w;
    float ss = v.x * v.x + v.y * v.y + v.z * v.z + v.w * v.w;
    for (int m = 16; m >= 1; m >>= 1) { s += __shfl_xor(s, m, 32); ss += __shfl_xor(ss, m, 32); }
    float mean = s * (1.f / IN_C);
    float var  = ss * (1.f / IN_C) - mean * mean;
    float rstd = rsqrtf(var + LN_EPS);
    int mt = wave >> 4, mrow = wave & 15;
    float vv[4] = {v.x, v.y, v.z, v.w};
    for (int i = 0; i < 4; ++i) {
        int k = lane * 4 + i;
        float hx = (vv[i] - mean) * rstd * g[k] + beta[k];
        int kt = k >> 5, kl = k & 31, l2, e2;
        a_pos(mrow, kl, l2, e2);
        size_t idx = ((size_t)(mt * 4 + kt)) * 512 + l2 * 16 + e2;
        hpack[idx] = f2bf(hx);
        xpack[idx] = f2bf(vv[i]);
    }
}

// ---------- 3) x_l = h@Wl^T, x_r = h@Wr^T, x_res = x@Wa1^T + b ----------
// A fragments (h + x for this node tile, 8KB) async-staged into LDS once,
// reused by all 8 waves x 36 tiles.
__global__ void k_gemm1(const unsigned short* __restrict__ hpack,
                        const unsigned short* __restrict__ xpack,
                        const unsigned short* __restrict__ wl,
                        const unsigned short* __restrict__ wr,
                        const unsigned short* __restrict__ wa1,
                        const float* __restrict__ ba1,
                        float* __restrict__ xl, float* __restrict__ xr,
                        float* __restrict__ xres) {
    __shared__ __align__(16) unsigned short smem[4096];  // [0,2048)=h, [2048,4096)=x
    int mt = blockIdx.x;
    int tid = threadIdx.x;
    int w = tid >> 5, lane = tid & 31;

    unsigned lds_base = (unsigned)(size_t)(&smem[0]);
    async_stage(hpack + (size_t)mt * 2048, lds_base,        256, tid, 256);
    async_stage(xpack + (size_t)mt * 2048, lds_base + 4096, 256, tid, 256);
    asm volatile("s_wait_asynccnt 0x0" ::: "memory");
    __syncthreads();

    for (int t = w; t < 36; t += 8) {
        const unsigned short* A = (t < 32) ? smem : smem + 2048;
        const unsigned short* B;
        float* out; int nt, ldn; bool addb = false;
        if (t < 16)      { nt = t;      B = wl  + (size_t)(nt * 4) * 512; out = xl;   ldn = HC; }
        else if (t < 32) { nt = t - 16; B = wr  + (size_t)(nt * 4) * 512; out = xr;   ldn = HC; }
        else             { nt = t - 32; B = wa1 + (size_t)(nt * 4) * 512; out = xres; ldn = OUT_C; addb = true; }
        v8f acc = {0.f, 0.f, 0.f, 0.f, 0.f, 0.f, 0.f, 0.f};
        for (int kt = 0; kt < 4; ++kt)
            acc = wmma_step(A + (size_t)kt * 512, B + (size_t)kt * 512, acc);
        int col = nt * 16 + (lane & 15);
        float bb = addb ? ba1[col] : 0.f;
        for (int v = 0; v < 8; ++v) {
            int row = mt * 16 + v + ((lane >> 4) << 3);
            out[(size_t)row * ldn + col] = acc[v] + bb;
        }
    }
}

// ---------- 4) init segment buffers ----------
__global__ void k_init(unsigned* __restrict__ mx, float* __restrict__ den,
                       float* __restrict__ agg, int N) {
    int t = blockIdx.x * blockDim.x + threadIdx.x;
    if (t < N * 4) { mx[t] = 0u; den[t] = 0.f; }
    int tot = N * HC, stride = gridDim.x * blockDim.x;
    for (int i = t; i < tot; i += stride) agg[i] = 0.f;
}

// ---------- 5) edge logits + segment max (ordered-uint atomicMax) ----------
__global__ void k_edge1(const int* __restrict__ ei, const float* __restrict__ xl,
                        const float* __restrict__ xr, const float* __restrict__ att,
                        float* __restrict__ logits, unsigned* __restrict__ mx, int E) {
    int wave = (blockIdx.x * blockDim.x + threadIdx.x) >> 5;
    int lane = threadIdx.x & 31;
    if (wave >= E) return;
    int dst = ei[wave], src = ei[E + wave];
    const float4* pd = reinterpret_cast<const float4*>(xl + (size_t)dst * HC + lane * 8);
    const float4* ps = reinterpret_cast<const float4*>(xr + (size_t)src * HC + lane * 8);
    const float4* pa = reinterpret_cast<const float4*>(att + lane * 8);
    float part = 0.f;
    for (int q = 0; q < 2; ++q) {
        float4 d = pd[q], s = ps[q], a = pa[q];
        float e;
        e = d.x + s.x; e = e > 0.f ? e : 0.2f * e; part += e * a.x;
        e = d.y + s.y; e = e > 0.f ? e : 0.2f * e; part += e * a.y;
        e = d.z + s.z; e = e > 0.f ? e : 0.2f * e; part += e * a.z;
        e = d.w + s.w; e = e > 0.f ? e : 0.2f * e; part += e * a.w;
    }
    part += __shfl_xor(part, 1, 32);
    part += __shfl_xor(part, 2, 32);
    part += __shfl_xor(part, 4, 32);
    if ((lane & 7) == 0) {
        int h = lane >> 3;
        logits[(size_t)wave * 4 + h] = part;
        unsigned u = __float_as_uint(part);
        unsigned enc = (u & 0x80000000u) ? ~u : (u | 0x80000000u);
        atomicMax(mx + (size_t)dst * 4 + h, enc);
    }
}

// ---------- 6) exp + segment sum ----------
__global__ void k_edge2(const int* __restrict__ ei, float* __restrict__ logits,
                        const unsigned* __restrict__ mx, float* __restrict__ den, int E) {
    int t = blockIdx.x * blockDim.x + threadIdx.x;
    if (t >= E * 4) return;
    int e = t >> 2, h = t & 3;
    int dst = ei[e];
    unsigned enc = mx[dst * 4 + h];
    unsigned u = (enc & 0x80000000u) ? (enc & 0x7FFFFFFFu) : ~enc;
    float ex = __expf(logits[t] - __uint_as_float(u));
    logits[t] = ex;
    atomicAdd(den + dst * 4 + h, ex);
}

// ---------- 7) alpha-weighted scatter add ----------
__global__ void k_edge3(const int* __restrict__ ei, const float* __restrict__ ex,
                        const float* __restrict__ den, const float* __restrict__ xr,
                        float* __restrict__ agg, int E) {
    int wave = (blockIdx.x * blockDim.x + threadIdx.x) >> 5;
    int lane = threadIdx.x & 31;
    if (wave >= E) return;
    int dst = ei[wave], src = ei[E + wave];
    int h = lane >> 3;
    float alpha = ex[(size_t)wave * 4 + h] / (den[dst * 4 + h] + 1e-16f);
    const float* s = xr + (size_t)src * HC + lane * 8;
    float* a = agg + (size_t)dst * HC + lane * 8;
    for (int i = 0; i < 8; ++i) atomicAdd(a + i, s[i] * alpha);
}

// ---------- 8) combine + LN2, pack h2 & out (K=64 -> KT=2) ----------
__global__ void k_combine(const float* __restrict__ agg, const float* __restrict__ bias,
                          const float* __restrict__ xres, const float* __restrict__ g2,
                          const float* __restrict__ beta2,
                          unsigned short* __restrict__ h2pack,
                          unsigned short* __restrict__ outpack, int N) {
    int wave = (blockIdx.x * blockDim.x + threadIdx.x) >> 5;
    int lane = threadIdx.x & 31;
    if (wave >= N) return;
    float o[2];
    for (int i = 0; i < 2; ++i) {
        int c = lane * 2 + i;
        const float* a = agg + (size_t)wave * HC;
        float s = a[c] + a[64 + c] + a[128 + c] + a[192 + c];
        float attn = s * 0.25f + bias[c];
        o[i] = SH * attn + SH * xres[(size_t)wave * OUT_C + c];
    }
    float s = o[0] + o[1], ss = o[0] * o[0] + o[1] * o[1];
    for (int m = 16; m >= 1; m >>= 1) { s += __shfl_xor(s, m, 32); ss += __shfl_xor(ss, m, 32); }
    float mean = s * (1.f / OUT_C);
    float var  = ss * (1.f / OUT_C) - mean * mean;
    float rstd = rsqrtf(var + LN_EPS);
    int mt = wave >> 4, mrow = wave & 15;
    for (int i = 0; i < 2; ++i) {
        int c = lane * 2 + i;
        float h2 = (o[i] - mean) * rstd * g2[c] + beta2[c];
        int kt = c >> 5, kl = c & 31, l2, e2;
        a_pos(mrow, kl, l2, e2);
        size_t idx = ((size_t)(mt * 2 + kt)) * 512 + l2 * 16 + e2;
        h2pack[idx]  = f2bf(h2);
        outpack[idx] = f2bf(o[i]);
    }
}

// ---------- 9) FFN1: hidden = relu(h2@W1^T + b1), packed straight into A frags ----------
__global__ void k_ffn1(const unsigned short* __restrict__ h2pack,
                       const unsigned short* __restrict__ w1,
                       const float* __restrict__ b1,
                       unsigned short* __restrict__ hidpack) {
    __shared__ __align__(16) unsigned short smem[1024];  // h2: 2 frags = 2KB
    int mt = blockIdx.x;
    int tid = threadIdx.x;
    int w = tid >> 5, lane = tid & 31;

    unsigned lds_base = (unsigned)(size_t)(&smem[0]);
    async_stage(h2pack + (size_t)mt * 1024, lds_base, 128, tid, 256);
    asm volatile("s_wait_asynccnt 0x0" ::: "memory");
    __syncthreads();

    for (int nt = w; nt < 16; nt += 8) {
        v8f acc = {0.f, 0.f, 0.f, 0.f, 0.f, 0.f, 0.f, 0.f};
        for (int kt = 0; kt < 2; ++kt)
            acc = wmma_step(smem + (size_t)kt * 512,
                            w1 + ((size_t)(nt * 2 + kt)) * 512, acc);
        int col = nt * 16 + (lane & 15);
        float bb = b1[col];
        int kt = col >> 5, kl = col & 31;
        for (int v = 0; v < 8; ++v) {
            int mrow = v + ((lane >> 4) << 3);
            float hv = acc[v] + bb;
            hv = hv > 0.f ? hv : 0.f;
            int l2, e2;
            a_pos(mrow, kl, l2, e2);
            hidpack[((size_t)(mt * 8 + kt)) * 512 + l2 * 16 + e2] = f2bf(hv);
        }
    }
}

// ---------- 10) final: sh*(hidden@W2^T + b2) + sh*(out@Wa2^T + ba2) ----------
__global__ void k_final(const unsigned short* __restrict__ hidpack,
                        const unsigned short* __restrict__ w2,
                        const float* __restrict__ b2,
                        const unsigned short* __restrict__ outpack,
                        const unsigned short* __restrict__ wa2,
                        const float* __restrict__ ba2,
                        float* __restrict__ out) {
    __shared__ __align__(16) unsigned short smem[5120];  // [0,4096)=hid 8 frags, [4096,5120)=out 2 frags
    int mt = blockIdx.x;
    int tid = threadIdx.x;
    int w = tid >> 5, lane = tid & 31;

    unsigned lds_base = (unsigned)(size_t)(&smem[0]);
    async_stage(hidpack + (size_t)mt * 4096, lds_base,        512, tid, 128);
    async_stage(outpack + (size_t)mt * 1024, lds_base + 8192, 128, tid, 128);
    asm volatile("s_wait_asynccnt 0x0" ::: "memory");
    __syncthreads();

    int nt = w;  // block of 128 = 4 waves, one 16-col tile each
    v8f accf = {0.f, 0.f, 0.f, 0.f, 0.f, 0.f, 0.f, 0.f};
    v8f accr = {0.f, 0.f, 0.f, 0.f, 0.f, 0.f, 0.f, 0.f};
    for (int kt = 0; kt < 8; ++kt)
        accf = wmma_step(smem + (size_t)kt * 512,
                         w2 + ((size_t)(nt * 8 + kt)) * 512, accf);
    for (int kt = 0; kt < 2; ++kt)
        accr = wmma_step(smem + 4096 + (size_t)kt * 512,
                         wa2 + ((size_t)(nt * 2 + kt)) * 512, accr);
    int col = nt * 16 + (lane & 15);
    float bf_ = b2[col], br = ba2[col];
    for (int v = 0; v < 8; ++v) {
        int row = mt * 16 + v + ((lane >> 4) << 3);
        out[(size_t)row * OUT_C + col] = SH * (accf[v] + bf_) + SH * (accr[v] + br);
    }
}

// ---------- host ----------
extern "C" void kernel_launch(void* const* d_in, const int* in_sizes, int n_in,
                              void* d_out, int out_size, void* d_ws, size_t ws_size,
                              hipStream_t stream) {
    const float* x    = (const float*)d_in[0];
    const int*   ei   = (const int*)d_in[1];
    const float* Wl   = (const float*)d_in[2];
    const float* Wr   = (const float*)d_in[3];
    const float* att  = (const float*)d_in[4];
    const float* bias = (const float*)d_in[5];
    const float* Wa1  = (const float*)d_in[6];
    const float* ba1  = (const float*)d_in[7];
    const float* Wa2  = (const float*)d_in[8];
    const float* ba2  = (const float*)d_in[9];
    const float* g1   = (const float*)d_in[10];
    const float* be1  = (const float*)d_in[11];
    const float* g2   = (const float*)d_in[12];
    const float* be2  = (const float*)d_in[13];
    const float* W1   = (const float*)d_in[14];
    const float* b1   = (const float*)d_in[15];
    const float* W2   = (const float*)d_in[16];
    const float* b2   = (const float*)d_in[17];
    float* out = (float*)d_out;

    const int N  = in_sizes[0] / IN_C;
    const int E  = in_sizes[1] / 2;
    const int MT = (N + 15) / 16;

    char* ws = (char*)d_ws;
    size_t off = 0;
    auto carve = [&](size_t bytes) -> char* {
        char* p = ws + off;
        off = (off + bytes + 255) & ~(size_t)255;
        return p;
    };
    unsigned short* wl_p  = (unsigned short*)carve(16 * 4 * 512 * 2);
    unsigned short* wr_p  = (unsigned short*)carve(16 * 4 * 512 * 2);
    unsigned short* wa1_p = (unsigned short*)carve(4  * 4 * 512 * 2);
    unsigned short* w1_p  = (unsigned short*)carve(16 * 2 * 512 * 2);
    unsigned short* w2_p  = (unsigned short*)carve(4  * 8 * 512 * 2);
    unsigned short* wa2_p = (unsigned short*)carve(4  * 2 * 512 * 2);
    unsigned short* x_p   = (unsigned short*)carve((size_t)MT * 4 * 512 * 2);
    unsigned short* h_p   = (unsigned short*)carve((size_t)MT * 4 * 512 * 2);
    float*          xl    = (float*)carve((size_t)N * HC * 4);
    float*          xr    = (float*)carve((size_t)N * HC * 4);
    float*          xres  = (float*)carve((size_t)N * OUT_C * 4);
    float*          lg    = (float*)carve((size_t)E * 4 * 4);
    unsigned*       mx    = (unsigned*)carve((size_t)N * 4 * 4);
    float*          den   = (float*)carve((size_t)N * 4 * 4);
    float*          agg   = (float*)carve((size_t)N * HC * 4);
    unsigned short* h2_p  = (unsigned short*)carve((size_t)MT * 2 * 512 * 2);
    unsigned short* o_p   = (unsigned short*)carve((size_t)MT * 2 * 512 * 2);
    unsigned short* hid_p = (unsigned short*)carve((size_t)MT * 8 * 512 * 2);
    (void)ws_size; (void)n_in; (void)out_size;

    // weight packing
    k_pack_w<<<(16 * 4 * 512 + 255) / 256, 256, 0, stream>>>(Wl,  wl_p,  16, 4, IN_C);
    k_pack_w<<<(16 * 4 * 512 + 255) / 256, 256, 0, stream>>>(Wr,  wr_p,  16, 4, IN_C);
    k_pack_w<<<(4  * 4 * 512 + 255) / 256, 256, 0, stream>>>(Wa1, wa1_p, 4,  4, IN_C);
    k_pack_w<<<(16 * 2 * 512 + 255) / 256, 256, 0, stream>>>(W1,  w1_p,  16, 2, OUT_C);
    k_pack_w<<<(4  * 8 * 512 + 255) / 256, 256, 0, stream>>>(W2,  w2_p,  4,  8, FFN_H);
    k_pack_w<<<(4  * 2 * 512 + 255) / 256, 256, 0, stream>>>(Wa2, wa2_p, 4,  2, OUT_C);

    // LN1 + pack
    k_ln1_pack<<<(N + 7) / 8, 256, 0, stream>>>(x, g1, be1, x_p, h_p, N);

    // projections (WMMA, LDS-staged A)
    k_gemm1<<<MT, 256, 0, stream>>>(h_p, x_p, wl_p, wr_p, wa1_p, ba1, xl, xr, xres);

    // segment buffers
    k_init<<<(N * 4 + 255) / 256, 256, 0, stream>>>(mx, den, agg, N);

    // edge passes
    k_edge1<<<(E + 7) / 8, 256, 0, stream>>>(ei, xl, xr, att, lg, mx, E);
    k_edge2<<<(E * 4 + 255) / 256, 256, 0, stream>>>(ei, lg, mx, den, E);
    k_edge3<<<(E + 7) / 8, 256, 0, stream>>>(ei, lg, den, xr, agg, E);

    // combine + LN2 + pack
    k_combine<<<(N + 7) / 8, 256, 0, stream>>>(agg, bias, xres, g2, be2, h2_p, o_p, N);

    // FFN (WMMA, LDS-staged A)
    k_ffn1<<<MT, 256, 0, stream>>>(h2_p, w1_p, b1, hid_p);
    k_final<<<MT, 128, 0, stream>>>(hid_p, w2_p, b2, o_p, wa2_p, ba2, out);
}
